// InBatchINFGRU_84413287236244
// MI455X (gfx1250) — compile-verified, matrix operand order checked by reading, not run on hardware
//
#include <hip/hip_runtime.h>
#include <stdint.h>

#define Bq 1024
#define Hd 1536
#define H3 (3 * Hd)
#define MAXNEG 32
#define TEMP_INV 20.0f   // 1 / 0.05

#define KT 32     // K slice per WMMA step
#define MT 64     // output rows per block
#define NTILE 64  // output cols per block
#define LDA 40    // LDS row stride in bf16 units (80 B: 16B-aligned, bank-spread)

typedef __attribute__((ext_vector_type(16))) __bf16 v16bf;
typedef __attribute__((ext_vector_type(8)))  float  v8f;

__device__ __forceinline__ unsigned short f2bf(float f) {
  union { float f; unsigned u; } v; v.f = f;
  unsigned r = v.u + 0x7FFFu + ((v.u >> 16) & 1u); // round-to-nearest-even
  return (unsigned short)(r >> 16);
}

#if defined(__has_builtin)
#if __has_builtin(__builtin_amdgcn_cvt_pk_bf16_f32)
#define HAVE_PK_BF16 1
#endif
#endif

__device__ __forceinline__ unsigned pack2(float lo, float hi) {
#ifdef HAVE_PK_BF16
  auto r = __builtin_amdgcn_cvt_pk_bf16_f32(lo, hi);
  unsigned u; __builtin_memcpy(&u, &r, sizeof(u));
  return u;
#else
  return (unsigned)f2bf(lo) | ((unsigned)f2bf(hi) << 16);
#endif
}
__device__ __forceinline__ float sigmf(float x) { return 1.0f / (1.0f + __expf(-x)); }

// CDNA5 async global->LDS copy (no VGPR round-trip, tracked by ASYNCcnt).
// LDS destination = low 32 bits of the generic shared-pointer address
// (ISA 10.2).
__device__ __forceinline__ void async_load_b128(unsigned lds_off, const void* gptr) {
  asm volatile("global_load_async_to_lds_b128 %0, %1, off"
               :: "v"(lds_off), "v"((unsigned long long)(uintptr_t)gptr)
               : "memory");
}
__device__ __forceinline__ void wait_async() {
  asm volatile("s_wait_asynccnt 0x0" ::: "memory");
}
__device__ __forceinline__ unsigned lds_off32(const void* p) {
  return (unsigned)(uintptr_t)p;
}

// One 16x32 bf16 A/B fragment = two 16-byte LDS loads (K-contiguous per half).
// CDNA5 ISA 7.12.2: lanes 0-15 hold K 0..7 / 16..23, lanes 16-31 hold K 8..15 / 24..31.
__device__ __forceinline__ v16bf load_frag(const unsigned short* rowptr, int lhalf) {
  union { v16bf v; uint4 q[2]; } f;
  f.q[0] = *(const uint4*)(rowptr + lhalf * 8);
  f.q[1] = *(const uint4*)(rowptr + 16 + lhalf * 8);
  return f.v;
}

__device__ __forceinline__ v8f bfwmma(v16bf a, v16bf b, v8f c) {
  return __builtin_amdgcn_wmma_f32_16x16x32_bf16(false, a, false, b, (short)0, c,
                                                 false, false);
}

// ---------------------------------------------------------------------------
__global__ __launch_bounds__(1024) void prefix_kernel(const int* __restrict__ lens,
                                                      int* __restrict__ offs) {
  __shared__ int s[Bq];
  int tid = threadIdx.x;
  s[tid] = lens[tid];
  __syncthreads();
  for (int d = 1; d < Bq; d <<= 1) {
    int v = (tid >= d) ? s[tid - d] : 0;
    __syncthreads();
    s[tid] += v;
    __syncthreads();
  }
  offs[tid] = s[tid] - lens[tid];
}

// 8 elements per thread (n must be a multiple of 8 — all our sizes are)
__global__ void cvt_bf16x8_kernel(const float* __restrict__ src,
                                  unsigned short* __restrict__ dst, int n8) {
  int i = blockIdx.x * blockDim.x + threadIdx.x;
  if (i < n8) {
    float4 f0 = *(const float4*)(src + (size_t)i * 8);
    float4 f1 = *(const float4*)(src + (size_t)i * 8 + 4);
    uint4 pk;
    pk.x = pack2(f0.x, f0.y);
    pk.y = pack2(f0.z, f0.w);
    pk.z = pack2(f1.x, f1.y);
    pk.w = pack2(f1.z, f1.w);
    *(uint4*)&dst[(size_t)i * 8] = pk;
  }
}

__global__ void init_h8_kernel(const float* __restrict__ q, float* __restrict__ hf,
                               unsigned short* __restrict__ hbf, int n8) {
  int i = blockIdx.x * blockDim.x + threadIdx.x;
  if (i < n8) {
    float4 f0 = *(const float4*)(q + (size_t)i * 8);
    float4 f1 = *(const float4*)(q + (size_t)i * 8 + 4);
    *(float4*)(hf + (size_t)i * 8)     = f0;
    *(float4*)(hf + (size_t)i * 8 + 4) = f1;
    uint4 pk;
    pk.x = pack2(f0.x, f0.y);
    pk.y = pack2(f0.z, f0.w);
    pk.z = pack2(f1.x, f1.y);
    pk.w = pack2(f1.z, f1.w);
    *(uint4*)&hbf[(size_t)i * 8] = pk;
  }
}

// ---------------------------------------------------------------------------
// One fused GRU step. LDS double-buffered with CDNA5 async global->LDS loads;
// K-loop unrolled by 2 so buffer selection is compile-time (immediate LDS
// offsets, no select chains). x-phase: planes {r, z, i_n} += x@Wih^T;
// h-phase: planes {r, z, h_n} += h@Whh^T (r/z share accumulators), then gates.
__global__ __launch_bounds__(256)
void gru_step_kernel(const unsigned short* __restrict__ xbf,  // [total, Hd] bf16
                     const int* __restrict__ offs, const int* __restrict__ lens,
                     const unsigned short* __restrict__ Wih,
                     const unsigned short* __restrict__ Whh,
                     const float* __restrict__ bih, const float* __restrict__ bhh,
                     const float* __restrict__ h_in_f,
                     const unsigned short* __restrict__ h_in_bf,
                     float* __restrict__ h_out_f,
                     unsigned short* __restrict__ h_out_bf,
                     int t, int total) {
  __shared__ __align__(16) unsigned short lds_a[2][MT * LDA];
  __shared__ __align__(16) unsigned short lds_w[2][3 * NTILE * LDA];

  const int tid   = threadIdx.x;
  const int lane  = tid & 31;
  const int w     = tid >> 5;
  const int wm    = w & 3;   // 4 wave-rows * 16 rows
  const int wn    = w >> 2;  // 2 wave-cols * 32 cols
  const int lhalf = lane >> 4;
  const int ln    = lane & 15;
  const int brow0 = blockIdx.x * MT;
  const int bcol0 = blockIdx.y * NTILE;

  // staging geometry (k0-invariant)
  const int srow = tid >> 2;        // staging row 0..63
  const int skc  = (tid & 3) * 8;   // staging k-chunk {0,8,16,24}
  const int adst = srow * LDA + skc;

  size_t wsrc[3];
  unsigned aoff[2], woff[2][3];
#pragma unroll
  for (int buf = 0; buf < 2; ++buf) aoff[buf] = lds_off32(&lds_a[buf][adst]);
#pragma unroll
  for (int it = 0; it < 3; ++it) {
    int c = tid + it * 256;  // 0..767 = 3 gates * 64 rows * 4 chunks
    int g = c >> 8, rem = c & 255, jl = rem >> 2, qd = rem & 3;
    wsrc[it] = (size_t)(g * Hd + bcol0 + jl) * Hd + qd * 8;
    int wd = (g * NTILE + jl) * LDA + qd * 8;
#pragma unroll
    for (int buf = 0; buf < 2; ++buf) woff[buf][it] = lds_off32(&lds_w[buf][wd]);
  }

  const int sb = brow0 + srow;
  int idx = offs[sb] + t;
  idx = (idx < total - 1) ? idx : (total - 1);
  const unsigned short* xsrc = xbf + (size_t)idx * Hd + skc;
  const unsigned short* hsrc = h_in_bf + (size_t)sb * Hd + skc;

  // planes: 0 = i_r+h_r, 1 = i_z+h_z, 2 = i_n, 3 = h_n ; 2 N-tiles each
  v8f zero = {};
  v8f acc[4][2];
#pragma unroll
  for (int p = 0; p < 4; ++p)
#pragma unroll
    for (int nt = 0; nt < 2; ++nt) acc[p][nt] = zero;

#pragma unroll
  for (int phase = 0; phase < 2; ++phase) {
    const unsigned short* W    = phase ? Whh : Wih;
    const unsigned short* asrc = phase ? hsrc : xsrc;
    const int nplane           = phase ? 3 : 2;

    auto stage_slice = [&](int buf, int k) {
      async_load_b128(aoff[buf], asrc + k);          // buf is compile-time
      async_load_b128(woff[buf][0], W + wsrc[0] + k);
      async_load_b128(woff[buf][1], W + wsrc[1] + k);
      async_load_b128(woff[buf][2], W + wsrc[2] + k);
    };
    auto compute_slice = [&](int buf) {
      const unsigned short* la = lds_a[buf] + (wm * 16 + ln) * LDA;
      const unsigned short* lw = lds_w[buf] + (wn * 32 + ln) * LDA;
      v16bf a   = load_frag(la, lhalf);
      v16bf br0 = load_frag(lw, lhalf);
      v16bf bz0 = load_frag(lw + NTILE * LDA, lhalf);
      v16bf bn0 = load_frag(lw + 2 * NTILE * LDA, lhalf);
      v16bf br1 = load_frag(lw + 16 * LDA, lhalf);
      v16bf bz1 = load_frag(lw + NTILE * LDA + 16 * LDA, lhalf);
      v16bf bn1 = load_frag(lw + 2 * NTILE * LDA + 16 * LDA, lhalf);
      acc[0][0]      = bfwmma(a, br0, acc[0][0]);
      acc[1][0]      = bfwmma(a, bz0, acc[1][0]);
      acc[nplane][0] = bfwmma(a, bn0, acc[nplane][0]);
      acc[0][1]      = bfwmma(a, br1, acc[0][1]);
      acc[1][1]      = bfwmma(a, bz1, acc[1][1]);
      acc[nplane][1] = bfwmma(a, bn1, acc[nplane][1]);
    };

    // prologue: stage K-slice 0 into buf0
    stage_slice(0, 0);
    wait_async();
    __syncthreads();

    // Hd/KT = 48 iterations, paired: even -> compute buf0 / stage buf1,
    // odd -> compute buf1 / stage buf0. All LDS offsets are immediates.
    for (int k0 = 0; k0 < Hd; k0 += 2 * KT) {
      // even half: compute k0 from buf0, stage k0+KT into buf1
      stage_slice(1, k0 + KT);
      if (k0 + 2 * KT < Hd)  // gfx1250 global_prefetch two slices ahead
        __builtin_prefetch(W + wsrc[0] + k0 + 2 * KT, 0, 1);
      compute_slice(0);
      wait_async();
      __syncthreads();

      // odd half: compute k0+KT from buf1, stage k0+2KT into buf0
      if (k0 + 2 * KT < Hd) stage_slice(0, k0 + 2 * KT);
      compute_slice(1);
      wait_async();
      __syncthreads();
    }
  }

  // ---- gate nonlinearity + masked update
  float rb[2], zb[2], inb[2], hnb[2];
#pragma unroll
  for (int nt = 0; nt < 2; ++nt) {
    int j   = bcol0 + wn * 32 + nt * 16 + ln;
    rb[nt]  = bih[j] + bhh[j];
    zb[nt]  = bih[Hd + j] + bhh[Hd + j];
    inb[nt] = bih[2 * Hd + j];
    hnb[nt] = bhh[2 * Hd + j];
  }
#pragma unroll
  for (int v = 0; v < 8; ++v) {
    int b = brow0 + wm * 16 + lhalf * 8 + v;  // C/D layout: M = v (+8 hi half)
    bool active = (t < lens[b]);
#pragma unroll
    for (int nt = 0; nt < 2; ++nt) {
      int j = bcol0 + wn * 32 + nt * 16 + ln;
      float hold = h_in_f[(size_t)b * Hd + j];
      float r  = sigmf(acc[0][nt][v] + rb[nt]);
      float z  = sigmf(acc[1][nt][v] + zb[nt]);
      float nn = tanhf(acc[2][nt][v] + inb[nt] + r * (acc[3][nt][v] + hnb[nt]));
      float hnew = (1.0f - z) * nn + z * hold;
      float hv = active ? hnew : hold;
      h_out_f[(size_t)b * Hd + j]  = hv;
      h_out_bf[(size_t)b * Hd + j] = f2bf(hv);
    }
  }
}

// ---------------------------------------------------------------------------
// scores = (h_final @ k^T) / TEMP — WMMA GEMM, async-staged, 2x-unrolled
__global__ __launch_bounds__(256)
void scores_kernel(const unsigned short* __restrict__ qbf,
                   const unsigned short* __restrict__ kbf,
                   float* __restrict__ scores) {
  __shared__ __align__(16) unsigned short lds_a[2][MT * LDA];
  __shared__ __align__(16) unsigned short lds_b[2][NTILE * LDA];

  const int tid   = threadIdx.x;
  const int lane  = tid & 31;
  const int w     = tid >> 5;
  const int wm    = w & 3;
  const int wn    = w >> 2;
  const int lhalf = lane >> 4;
  const int ln    = lane & 15;
  const int brow0 = blockIdx.x * MT;
  const int bcol0 = blockIdx.y * NTILE;

  const int srow = tid >> 2;
  const int skc  = (tid & 3) * 8;
  const int adst = srow * LDA + skc;

  unsigned aoff[2], boff[2];
#pragma unroll
  for (int buf = 0; buf < 2; ++buf) {
    aoff[buf] = lds_off32(&lds_a[buf][adst]);
    boff[buf] = lds_off32(&lds_b[buf][adst]);
  }

  const unsigned short* qsrc = qbf + (size_t)(brow0 + srow) * Hd + skc;
  const unsigned short* ksrc = kbf + (size_t)(bcol0 + srow) * Hd + skc;

  v8f acc0 = {}, acc1 = {};

  auto stage_slice = [&](int buf, int k) {
    async_load_b128(aoff[buf], qsrc + k);
    async_load_b128(boff[buf], ksrc + k);
  };
  auto compute_slice = [&](int buf) {
    v16bf a  = load_frag(lds_a[buf] + (wm * 16 + ln) * LDA, lhalf);
    v16bf b0 = load_frag(lds_b[buf] + (wn * 32 + ln) * LDA, lhalf);
    v16bf b1 = load_frag(lds_b[buf] + (wn * 32 + 16 + ln) * LDA, lhalf);
    acc0 = bfwmma(a, b0, acc0);
    acc1 = bfwmma(a, b1, acc1);
  };

  stage_slice(0, 0);
  wait_async();
  __syncthreads();

  for (int k0 = 0; k0 < Hd; k0 += 2 * KT) {
    stage_slice(1, k0 + KT);
    compute_slice(0);
    wait_async();
    __syncthreads();

    if (k0 + 2 * KT < Hd) stage_slice(0, k0 + 2 * KT);
    compute_slice(1);
    wait_async();
    __syncthreads();
  }

#pragma unroll
  for (int nt = 0; nt < 2; ++nt) {
    int j = bcol0 + wn * 32 + nt * 16 + ln;
#pragma unroll
    for (int v = 0; v < 8; ++v) {
      int b = brow0 + wm * 16 + lhalf * 8 + v;
      float val = (nt == 0 ? acc0[v] : acc1[v]) * TEMP_INV;
      scores[(size_t)b * Bq + j] = val;
    }
  }
}

// ---------------------------------------------------------------------------
__global__ __launch_bounds__(256)
void rowloss_kernel(const float* __restrict__ S, float* __restrict__ term) {
  __shared__ float red[256];
  int i = blockIdx.x;
  const float* row = S + (size_t)i * Bq;
  float mx = -1e30f;
  for (int c = threadIdx.x; c < Bq; c += 256) mx = fmaxf(mx, row[c]);
  red[threadIdx.x] = mx; __syncthreads();
  for (int s = 128; s > 0; s >>= 1) {
    if (threadIdx.x < s) red[threadIdx.x] = fmaxf(red[threadIdx.x], red[threadIdx.x + s]);
    __syncthreads();
  }
  mx = red[0]; __syncthreads();
  float se = 0.f;
  for (int c = threadIdx.x; c < Bq; c += 256) se += __expf(row[c] - mx);
  red[threadIdx.x] = se; __syncthreads();
  for (int s = 128; s > 0; s >>= 1) {
    if (threadIdx.x < s) red[threadIdx.x] += red[threadIdx.x + s];
    __syncthreads();
  }
  if (threadIdx.x == 0) term[i] = -(row[i] - mx - __logf(red[0]));
}

__global__ __launch_bounds__(256)
void loss_reduce_kernel(const float* __restrict__ term, float* __restrict__ out) {
  __shared__ float red[256];
  float s = 0.f;
  for (int i = threadIdx.x; i < Bq; i += 256) s += term[i];
  red[threadIdx.x] = s; __syncthreads();
  for (int st = 128; st > 0; st >>= 1) {
    if (threadIdx.x < st) red[threadIdx.x] += red[threadIdx.x + st];
    __syncthreads();
  }
  if (threadIdx.x == 0) out[0] = red[0] / (float)Bq;
}

// ---------------------------------------------------------------------------
extern "C" void kernel_launch(void* const* d_in, const int* in_sizes, int n_in,
                              void* d_out, int out_size, void* d_ws, size_t ws_size,
                              hipStream_t stream) {
  const float* q    = (const float*)d_in[0];
  const float* kemb = (const float*)d_in[1];
  const float* inpn = (const float*)d_in[2];
  const int*   lens = (const int*)d_in[3];
  const float* Wih  = (const float*)d_in[4];
  const float* Whh  = (const float*)d_in[5];
  const float* bih  = (const float*)d_in[6];
  const float* bhh  = (const float*)d_in[7];
  float* out = (float*)d_out;
  int total = in_sizes[2] / Hd;

  char* p = (char*)d_ws;
  auto alloc = [&](size_t bytes) -> char* {
    char* r = p;
    p += (bytes + 255) & ~(size_t)255;
    return r;
  };
  int* offs               = (int*)alloc((size_t)Bq * 4);
  unsigned short* Wih_bf  = (unsigned short*)alloc((size_t)H3 * Hd * 2);
  unsigned short* Whh_bf  = (unsigned short*)alloc((size_t)H3 * Hd * 2);
  unsigned short* kemb_bf = (unsigned short*)alloc((size_t)Bq * Hd * 2);
  unsigned short* x_bf    = (unsigned short*)alloc((size_t)total * Hd * 2);
  float* hf0              = (float*)alloc((size_t)Bq * Hd * 4);
  float* hf1              = (float*)alloc((size_t)Bq * Hd * 4);
  unsigned short* hb0     = (unsigned short*)alloc((size_t)Bq * Hd * 2);
  unsigned short* hb1     = (unsigned short*)alloc((size_t)Bq * Hd * 2);
  float* scores           = (float*)alloc((size_t)Bq * Bq * 4);
  float* term             = (float*)alloc((size_t)Bq * 4);
  float* hf[2]            = {hf0, hf1};
  unsigned short* hbf[2]  = {hb0, hb1};

  prefix_kernel<<<1, 1024, 0, stream>>>(lens, offs);

  int nW8 = (H3 * Hd) / 8;
  cvt_bf16x8_kernel<<<(nW8 + 255) / 256, 256, 0, stream>>>(Wih, Wih_bf, nW8);
  cvt_bf16x8_kernel<<<(nW8 + 255) / 256, 256, 0, stream>>>(Whh, Whh_bf, nW8);
  int nK8 = (Bq * Hd) / 8;
  cvt_bf16x8_kernel<<<(nK8 + 255) / 256, 256, 0, stream>>>(kemb, kemb_bf, nK8);
  int nX8 = (total * Hd) / 8;
  cvt_bf16x8_kernel<<<(nX8 + 255) / 256, 256, 0, stream>>>(inpn, x_bf, nX8);
  init_h8_kernel<<<(nK8 + 255) / 256, 256, 0, stream>>>(q, hf[0], hbf[0], nK8);

  dim3 sgrid(Bq / MT, Hd / NTILE);
  int cur = 0;
  for (int t = 0; t < MAXNEG; ++t) {
    gru_step_kernel<<<sgrid, 256, 0, stream>>>(x_bf, offs, lens, Wih_bf, Whh_bf,
                                               bih, bhh, hf[cur], hbf[cur],
                                               hf[1 - cur], hbf[1 - cur], t, total);
    cur = 1 - cur;
  }

  scores_kernel<<<dim3(Bq / MT, Bq / NTILE), 256, 0, stream>>>(hbf[cur], kemb_bf, scores);
  rowloss_kernel<<<Bq, 256, 0, stream>>>(scores, term);
  loss_reduce_kernel<<<1, 256, 0, stream>>>(term, out);
}
